// R2LEncoderBlock_56891136803415
// MI455X (gfx1250) — matrix operand order repaired and checked by reading, not compiled, not run on hardware
//
#include <hip/hip_runtime.h>
#include <math.h>

// ---------------- CDNA5 WMMA types ----------------
typedef __attribute__((ext_vector_type(16))) __bf16 v16bf;
typedef __attribute__((ext_vector_type(8)))  float  v8f;

#define DIMC   256
#define HEADS  8
#define HDIM   32
#define NTOK   50       // 1 regional + 49 window tokens
#define BATCH  32
#define NWIN   64       // windows per image (8x8)
#define NBLK   2048     // BATCH * NWIN
#define HID    1024
#define ATT_SCALE 0.176776695296636881f   // (256/8)^-0.5
#define LN_EPS 1e-5f

// float -> bf16 (round to nearest even); co-executes with XDL WMMA
__device__ inline unsigned short f2bf(float f) {
    unsigned int u = __float_as_uint(f);
    unsigned int r = u + 0x7FFFu + ((u >> 16) & 1u);
    return (unsigned short)(r >> 16);
}

// Swizzled-B tile: 32(K) x 16(N) bf16 stored so WMMA lane L reads its 16
// elements contiguously: element (k,n) at [ (k>>4)*16 + n ][ k&15 ]
__device__ inline int bswz(int k, int n) { return (((k >> 4) * 16 + n) << 4) + (k & 15); }

// B fragment from a swizzled tile: two 128-bit loads per lane.
__device__ inline v16bf load_b_swz(const unsigned short* tile) {
    int lane = threadIdx.x & 31;
    union { v16bf v; uint4 q[2]; } u;
    const uint4* p = (const uint4*)(tile + lane * 16);
    u.q[0] = p[0];
    u.q[1] = p[1];
    return u.v;
}

// A fragment: 16x32 bf16 from row-major source (leading dim ld, elements).
__device__ inline v16bf load_a_frag(const unsigned short* base, int ld) {
    int lane = threadIdx.x & 31;
    int half = lane >> 4, m = lane & 15;
    union { v16bf v; uint4 q[2]; } u;
    const unsigned short* p = base + m * ld + 16 * half;
    u.q[0] = *(const uint4*)(p);
    u.q[1] = *(const uint4*)(p + 8);
    return u.v;
}

__device__ inline v8f wmma_bf16(v16bf a, v16bf b, v8f c) {
    return __builtin_amdgcn_wmma_f32_16x16x32_bf16(false, a, false, b, (short)0, c,
                                                   false, false);
}

// LayerNorm of one row of tS (fp32 [rows][256]) into xln (bf16 [64][256])
__device__ inline void ln_row(const float* tS, unsigned short* xln, int r, int nvalid,
                              const float* g, const float* bta) {
    if (r < nvalid) {
        float sum = 0.f, sq = 0.f;
        for (int j = 0; j < DIMC; ++j) { float v = tS[r * DIMC + j]; sum += v; sq += v * v; }
        float mean = sum * (1.0f / DIMC);
        float var  = sq * (1.0f / DIMC) - mean * mean;
        float inv  = rsqrtf(var + LN_EPS);
        for (int j = 0; j < DIMC; ++j)
            xln[r * DIMC + j] = f2bf((tS[r * DIMC + j] - mean) * inv * g[j] + bta[j]);
    } else {
        for (int j = 0; j < DIMC; ++j) xln[r * DIMC + j] = 0;
    }
}

// Multi-head attention core: xln (bf16 [64][256] LDS) -> ao (bf16 [64][256] LDS)
template <int NVALID, bool HASBIAS>
__device__ void mhsa_core(const unsigned short* xln,
                          unsigned short* qs, unsigned short* ks, unsigned short* vs,
                          float* Ss, unsigned short* Ps, unsigned short* ao,
                          const unsigned short* bqkv, const float* qkvb,
                          const float* biasmat) {
    int tid = threadIdx.x;
    int w = tid >> 5, lane = tid & 31;
    int half = lane >> 4, nn = lane & 15;

    for (int h = 0; h < HEADS; ++h) {
        // ---- QKV: one (mtile,ntile) tile of q, k, v per wave.
        //      A-frag shared by 3 independent accumulator chains (no WMMA NOPs).
        {
            int mtile = w >> 1, ntile = w & 1;
            int ntq = (0 * DIMC + h * HDIM + ntile * 16) >> 4;
            int ntk = (1 * DIMC + h * HDIM + ntile * 16) >> 4;
            int ntv = (2 * DIMC + h * HDIM + ntile * 16) >> 4;
            v8f aq, ak, av;
#pragma unroll
            for (int r = 0; r < 8; ++r) { aq[r] = 0.f; ak[r] = 0.f; av[r] = 0.f; }
            for (int kk = 0; kk < 8; ++kk) {
                v16bf a  = load_a_frag(xln + (mtile * 16) * DIMC + kk * 32, DIMC);
                v16bf bq = load_b_swz(bqkv + (ntq * 8 + kk) * 512);
                v16bf bk = load_b_swz(bqkv + (ntk * 8 + kk) * 512);
                v16bf bv = load_b_swz(bqkv + (ntv * 8 + kk) * 512);
                aq = wmma_bf16(a, bq, aq);
                ak = wmma_bf16(a, bk, ak);
                av = wmma_bf16(a, bv, av);
            }
#pragma unroll
            for (int r = 0; r < 8; ++r) {
                int row = mtile * 16 + 8 * half + r;   // token index
                int col = ntile * 16 + nn;             // head-dim index
                qs[row * HDIM + col] = f2bf((aq[r] + qkvb[ntq * 16 + nn]) * ATT_SCALE);
                // k as B of q@k^T: B(k=col, n=row&15), key-tile = row>>4
                ks[(row >> 4) * 512 + bswz(col, row & 15)] =
                    f2bf(ak[r] + qkvb[ntk * 16 + nn]);
                // v as B of P@V: tile = (col>>4)*2 + (row>>5)
                vs[(((col >> 4) * 2) + (row >> 5)) * 512 + bswz(row & 31, nn)] =
                    f2bf(av[r] + qkvb[ntv * 16 + nn]);
            }
        }
        __syncthreads();
        // ---- S = q @ k^T (+ bias) : 16 tiles, 2 independent per wave ----
        {
            v16bf a0 = load_a_frag(qs + ((w + 0) >> 2) * 16 * HDIM, HDIM);
            v16bf b0 = load_b_swz(ks + ((w + 0) & 3) * 512);
            v16bf a1 = load_a_frag(qs + ((w + 8) >> 2) * 16 * HDIM, HDIM);
            v16bf b1 = load_b_swz(ks + ((w + 8) & 3) * 512);
            v8f s0, s1;
#pragma unroll
            for (int r = 0; r < 8; ++r) { s0[r] = 0.f; s1[r] = 0.f; }
            s0 = wmma_bf16(a0, b0, s0);
            s1 = wmma_bf16(a1, b1, s1);
#pragma unroll
            for (int j = 0; j < 2; ++j) {
                int tt = w + 8 * j;
                int mtile = tt >> 2, ntile = tt & 3;
                v8f acc = j ? s1 : s0;
#pragma unroll
                for (int r = 0; r < 8; ++r) {
                    int row = mtile * 16 + 8 * half + r;
                    int col = ntile * 16 + nn;
                    float val = acc[r];
                    if (HASBIAS && row < NTOK && col < NTOK)
                        val += biasmat[h * NTOK * NTOK + row * NTOK + col];
                    Ss[row * 64 + col] = val;
                }
            }
        }
        __syncthreads();
        // ---- softmax per row -> P (bf16) ----
        if (tid < 64) {
            int r = tid;
            if (r < NVALID) {
                float mx = -1e30f;
                for (int j2 = 0; j2 < NVALID; ++j2) mx = fmaxf(mx, Ss[r * 64 + j2]);
                float sum = 0.f;
                for (int j2 = 0; j2 < NVALID; ++j2) sum += __expf(Ss[r * 64 + j2] - mx);
                float inv = 1.f / sum;
                for (int j2 = 0; j2 < 64; ++j2)
                    Ps[r * 64 + j2] = (j2 < NVALID)
                        ? f2bf(__expf(Ss[r * 64 + j2] - mx) * inv) : (unsigned short)0;
            } else {
                for (int j2 = 0; j2 < 64; ++j2) Ps[r * 64 + j2] = 0;
            }
        }
        __syncthreads();
        // ---- O = P @ V -> ao columns [h*32, h*32+32) ----
        {
            int mtile = w >> 1, ntile = w & 1;
            v8f acc;
#pragma unroll
            for (int r = 0; r < 8; ++r) acc[r] = 0.f;
            for (int kk = 0; kk < 2; ++kk) {
                v16bf a = load_a_frag(Ps + mtile * 16 * 64 + kk * 32, 64);
                v16bf b = load_b_swz(vs + (ntile * 2 + kk) * 512);
                acc = wmma_bf16(a, b, acc);
            }
#pragma unroll
            for (int r = 0; r < 8; ++r) {
                int row = mtile * 16 + 8 * half + r;
                int col = h * HDIM + ntile * 16 + nn;
                ao[row * DIMC + col] = f2bf(acc[r]);
            }
        }
        __syncthreads();
    }
}

// GEMM with 8 output tiles per wave over K=256 (A: [64][256] bf16 LDS,
// B: swizzled weights, KT=8). acc[j]: mtile = j>>1, ntile = (j&1)? w+8 : w.
__device__ inline void gemm8_256(const unsigned short* A, const unsigned short* Bw,
                                 v8f acc[8], int w) {
    for (int kk = 0; kk < 8; ++kk) {
        v16bf a0 = load_a_frag(A + 0 * 16 * DIMC + kk * 32, DIMC);
        v16bf a1 = load_a_frag(A + 1 * 16 * DIMC + kk * 32, DIMC);
        v16bf a2 = load_a_frag(A + 2 * 16 * DIMC + kk * 32, DIMC);
        v16bf a3 = load_a_frag(A + 3 * 16 * DIMC + kk * 32, DIMC);
        v16bf b0 = load_b_swz(Bw + ((w + 0) * 8 + kk) * 512);
        v16bf b1 = load_b_swz(Bw + ((w + 8) * 8 + kk) * 512);
        acc[0] = wmma_bf16(a0, b0, acc[0]);
        acc[1] = wmma_bf16(a0, b1, acc[1]);
        acc[2] = wmma_bf16(a1, b0, acc[2]);
        acc[3] = wmma_bf16(a1, b1, acc[3]);
        acc[4] = wmma_bf16(a2, b0, acc[4]);
        acc[5] = wmma_bf16(a2, b1, acc[5]);
        acc[6] = wmma_bf16(a3, b0, acc[6]);
        acc[7] = wmma_bf16(a3, b1, acc[7]);
    }
}

// ---------------- Prep: bf16 swizzled weights + padded bias matrix ----------------
// tile (nt, kt) at offset ((nt * KT) + kt) * 512
__global__ void prep_kernel(const float* qkv_w, const float* proj_w,
                            const float* fc1_w, const float* fc2_w,
                            const float* bias_table, const int* rel_index,
                            unsigned short* bqkv, unsigned short* bproj,
                            unsigned short* b1w, unsigned short* b2w,
                            float* biasmat) {
    int idx = blockIdx.x * blockDim.x + threadIdx.x;
    int stride = gridDim.x * blockDim.x;
    for (int i = idx; i < 768 * 256; i += stride) {         // qkv: KT=8
        int o = i / 256, k = i % 256;
        bqkv[((o >> 4) * 8 + (k >> 5)) * 512 + bswz(k & 31, o & 15)] =
            f2bf(qkv_w[o * 256 + k]);
    }
    for (int i = idx; i < 256 * 256; i += stride) {         // proj: KT=8
        int o = i / 256, k = i % 256;
        bproj[((o >> 4) * 8 + (k >> 5)) * 512 + bswz(k & 31, o & 15)] =
            f2bf(proj_w[o * 256 + k]);
    }
    for (int i = idx; i < 1024 * 256; i += stride) {        // fc1: KT=8
        int o = i / 256, k = i % 256;
        b1w[((o >> 4) * 8 + (k >> 5)) * 512 + bswz(k & 31, o & 15)] =
            f2bf(fc1_w[o * 256 + k]);
    }
    for (int i = idx; i < 256 * 1024; i += stride) {        // fc2: KT=32
        int o = i / 1024, k = i % 1024;
        b2w[((o >> 4) * 32 + (k >> 5)) * 512 + bswz(k & 31, o & 15)] =
            f2bf(fc2_w[o * 1024 + k]);
    }
    for (int i = idx; i < HEADS * NTOK * NTOK; i += stride) {
        int h = i / (NTOK * NTOK), r = i % (NTOK * NTOK);
        int qi = r / NTOK, qj = r % NTOK;
        float v = 0.f;
        if (qi > 0 && qj > 0)
            v = bias_table[rel_index[(qi - 1) * 49 + (qj - 1)] * HEADS + h];
        biasmat[i] = v;
    }
}

// ---------------- RSA: attention over regional tokens ----------------
__global__ void __launch_bounds__(256)
rsa_kernel(const float* regional, const float* n1g, const float* n1b,
           const unsigned short* bqkv, const float* qkvb,
           const unsigned short* bproj, const float* projb,
           float* reg_upd) {
    __shared__ alignas(16) unsigned short xln[64 * DIMC];
    __shared__ alignas(16) unsigned short qs[64 * HDIM];
    __shared__ alignas(16) unsigned short ks[4 * 512];
    __shared__ alignas(16) unsigned short vs[4 * 512];
    __shared__ alignas(16) float Ss[64 * 64];
    __shared__ alignas(16) unsigned short Ps[64 * 64];
    __shared__ alignas(16) unsigned short ao[64 * DIMC];

    int bidx = blockIdx.x;
    int tid = threadIdx.x;
    const float* src = regional + (long)bidx * 64 * DIMC;

    if (tid < 64) {
        int r = tid;
        float sum = 0.f, sq = 0.f;
        for (int j = 0; j < DIMC; ++j) { float v = src[r * DIMC + j]; sum += v; sq += v * v; }
        float mean = sum * (1.0f / DIMC);
        float var  = sq * (1.0f / DIMC) - mean * mean;
        float inv  = rsqrtf(var + LN_EPS);
        for (int j = 0; j < DIMC; ++j)
            xln[r * DIMC + j] = f2bf((src[r * DIMC + j] - mean) * inv * n1g[j] + n1b[j]);
    }
    __syncthreads();

    mhsa_core<64, false>(xln, qs, ks, vs, Ss, Ps, ao, bqkv, qkvb, nullptr);

    // proj + residual -> reg_upd
    int w = tid >> 5, lane = tid & 31, half = lane >> 4, nn = lane & 15;
    v8f acc[8];
#pragma unroll
    for (int j = 0; j < 8; ++j)
        for (int r = 0; r < 8; ++r) acc[j][r] = 0.f;
    gemm8_256(ao, bproj, acc, w);
#pragma unroll
    for (int j = 0; j < 8; ++j) {
        int mtile = j >> 1, ntile = (j & 1) ? (w + 8) : w;
#pragma unroll
        for (int r = 0; r < 8; ++r) {
            int row = mtile * 16 + 8 * half + r;
            int col = ntile * 16 + nn;
            reg_upd[(long)bidx * 64 * DIMC + row * DIMC + col] =
                src[row * DIMC + col] + acc[j][r] + projb[col];
        }
    }
}

// ---------------- LSA + MLP fused: one block per window ----------------
__global__ void __launch_bounds__(256)
lsa_kernel(const float* x, const float* reg_upd,
           const float* n2g, const float* n2b, const float* n3g, const float* n3b,
           const unsigned short* bqkv, const float* qkvb,
           const unsigned short* bproj, const float* projb,
           const unsigned short* b1w, const float* b1b,
           const unsigned short* b2w, const float* b2b,
           const float* biasmat, float* out) {
    // 150 KB LDS, phase-overlapped (CDNA5 WGP has 320 KB)
    __shared__ alignas(16) unsigned char smem[153600];
    float*          tS  = (float*)smem;                             // [50][256] f32
    unsigned short* xln = (unsigned short*)(smem + 51200);          // [64][256] bf16
    unsigned short* ao  = (unsigned short*)(smem + 83968);          // [64][256] bf16
    unsigned char*  scr = smem + 116736;                            // 36864 B scratch
    unsigned short* qs  = (unsigned short*)scr;                     // [64][32]
    unsigned short* ks  = (unsigned short*)(scr + 4096);            // 4 swz tiles
    unsigned short* vs  = (unsigned short*)(scr + 8192);            // 4 swz tiles
    float*          Ss  = (float*)(scr + 12288);                    // [64][64] f32
    unsigned short* Ps  = (unsigned short*)(scr + 28672);           // [64][64] bf16
    unsigned short* ht  = (unsigned short*)scr;                     // [64][128] (MLP reuse)

    int i = blockIdx.x, tid = threadIdx.x;
    int b   = i & 31, nw = i >> 5;                 // regional-token indexing (window-major)
    int b2i = i >> 6, rem = i & 63;                // window indexing (batch-major)
    int wh = rem >> 3, ww = rem & 7;

    // ---- build t tile with 128-bit coalesced moves ----
    {
        float4* t4 = (float4*)tS;
        const float4* reg4 = (const float4*)(reg_upd + ((long)b * 64 + nw) * DIMC);
        for (int e4 = tid; e4 < NTOK * (DIMC / 4); e4 += 256) {
            int row = e4 >> 6, c4 = e4 & 63;
            float4 v;
            if (row == 0) {
                v = reg4[c4];
            } else {
                int r = row - 1, rr = r / 7, cc = r % 7;
                int pos = (wh * 7 + rr) * 56 + (ww * 7 + cc);
                v = *(const float4*)(x + ((long)b2i * 3136 + pos) * DIMC + c4 * 4);
            }
            t4[e4] = v;
        }
    }
    __syncthreads();

    if (tid < 64) ln_row(tS, xln, tid, NTOK, n2g, n2b);
    __syncthreads();

    mhsa_core<NTOK, true>(xln, qs, ks, vs, Ss, Ps, ao, bqkv, qkvb, biasmat);

    // ---- proj + residual into tS ----
    int w = tid >> 5, lane = tid & 31, half = lane >> 4, nn = lane & 15;
    {
        v8f acc[8];
#pragma unroll
        for (int j = 0; j < 8; ++j)
            for (int r = 0; r < 8; ++r) acc[j][r] = 0.f;
        gemm8_256(ao, bproj, acc, w);
#pragma unroll
        for (int j = 0; j < 8; ++j) {
            int mtile = j >> 1, ntile = (j & 1) ? (w + 8) : w;
#pragma unroll
            for (int r = 0; r < 8; ++r) {
                int row = mtile * 16 + 8 * half + r;
                int col = ntile * 16 + nn;
                if (row < NTOK) tS[row * DIMC + col] += acc[j][r] + projb[col];
            }
        }
    }
    __syncthreads();

    if (tid < 64) ln_row(tS, xln, tid, NTOK, n3g, n3b);
    __syncthreads();

    // ---- MLP: hidden streamed in 8 chunks of 128; FC2 kept in VGPRs ----
    v8f accG[8];
#pragma unroll
    for (int j = 0; j < 8; ++j)
        for (int r = 0; r < 8; ++r) accG[j][r] = 0.f;

    for (int c = 0; c < 8; ++c) {
        // hidden chunk: 4 tiles per wave (mtile=j, ntile=w), 1 B-frag per K-step
        {
            v8f hacc[4];
#pragma unroll
            for (int j = 0; j < 4; ++j)
                for (int r = 0; r < 8; ++r) hacc[j][r] = 0.f;
            int ntG = c * 8 + w;                       // global n-tile in [0,64)
            for (int kk = 0; kk < 8; ++kk) {
                v16bf bb = load_b_swz(b1w + (ntG * 8 + kk) * 512);
                v16bf a0 = load_a_frag(xln + 0 * 16 * DIMC + kk * 32, DIMC);
                v16bf a1 = load_a_frag(xln + 1 * 16 * DIMC + kk * 32, DIMC);
                v16bf a2 = load_a_frag(xln + 2 * 16 * DIMC + kk * 32, DIMC);
                v16bf a3 = load_a_frag(xln + 3 * 16 * DIMC + kk * 32, DIMC);
                hacc[0] = wmma_bf16(a0, bb, hacc[0]);
                hacc[1] = wmma_bf16(a1, bb, hacc[1]);
                hacc[2] = wmma_bf16(a2, bb, hacc[2]);
                hacc[3] = wmma_bf16(a3, bb, hacc[3]);
            }
#pragma unroll
            for (int j = 0; j < 4; ++j) {
#pragma unroll
                for (int r = 0; r < 8; ++r) {
                    int row = j * 16 + 8 * half + r;
                    int col = w * 16 + nn;
                    float v = hacc[j][r] + b1b[ntG * 16 + nn];
                    v = 0.5f * v * (1.0f + erff(v * 0.70710678118654752f)); // exact GELU
                    ht[row * 128 + col] = f2bf(v);
                }
            }
        }
        __syncthreads();
        // FC2 partial: 8 independent tiles per wave, 4 K-steps
        for (int kk = 0; kk < 4; ++kk) {
            v16bf a0 = load_a_frag(ht + 0 * 16 * 128 + kk * 32, 128);
            v16bf a1 = load_a_frag(ht + 1 * 16 * 128 + kk * 32, 128);
            v16bf a2 = load_a_frag(ht + 2 * 16 * 128 + kk * 32, 128);
            v16bf a3 = load_a_frag(ht + 3 * 16 * 128 + kk * 32, 128);
            v16bf b0 = load_b_swz(b2w + ((w + 0) * 32 + c * 4 + kk) * 512);
            v16bf b1 = load_b_swz(b2w + ((w + 8) * 32 + c * 4 + kk) * 512);
            accG[0] = wmma_bf16(a0, b0, accG[0]);
            accG[1] = wmma_bf16(a0, b1, accG[1]);
            accG[2] = wmma_bf16(a1, b0, accG[2]);
            accG[3] = wmma_bf16(a1, b1, accG[3]);
            accG[4] = wmma_bf16(a2, b0, accG[4]);
            accG[5] = wmma_bf16(a2, b1, accG[5]);
            accG[6] = wmma_bf16(a3, b0, accG[6]);
            accG[7] = wmma_bf16(a3, b1, accG[7]);
        }
        __syncthreads();
    }

    // ---- final: + b2 + residual; scatter to reg_out / x_out ----
    float* xout = out + (long)BATCH * NWIN * DIMC;
#pragma unroll
    for (int j = 0; j < 8; ++j) {
        int mtile = j >> 1, ntile = (j & 1) ? (w + 8) : w;
#pragma unroll
        for (int r = 0; r < 8; ++r) {
            int row = mtile * 16 + 8 * half + r;
            int col = ntile * 16 + nn;
            if (row >= NTOK) continue;
            float val = accG[j][r] + b2b[col] + tS[row * DIMC + col];
            if (row == 0) {
                out[((long)b * NWIN + nw) * DIMC + col] = val;   // reg_out[b][nw]
            } else {
                int rl = row - 1, rr = rl / 7, cc = rl % 7;
                int pos = (wh * 7 + rr) * 56 + (ww * 7 + cc);
                xout[((long)b2i * 3136 + pos) * DIMC + col] = val;
            }
        }
    }
}

// ---------------- launch ----------------
extern "C" void kernel_launch(void* const* d_in, const int* in_sizes, int n_in,
                              void* d_out, int out_size, void* d_ws, size_t ws_size,
                              hipStream_t stream) {
    (void)in_sizes; (void)n_in; (void)out_size; (void)ws_size;
    const float* regional  = (const float*)d_in[0];
    const float* x         = (const float*)d_in[1];
    const float* n1g       = (const float*)d_in[2];
    const float* n1b       = (const float*)d_in[3];
    const float* n2g       = (const float*)d_in[4];
    const float* n2b       = (const float*)d_in[5];
    const float* n3g       = (const float*)d_in[6];
    const float* n3b       = (const float*)d_in[7];
    const float* qkv_w     = (const float*)d_in[8];
    const float* qkv_b     = (const float*)d_in[9];
    const float* proj_w    = (const float*)d_in[10];
    const float* proj_b    = (const float*)d_in[11];
    const float* bias_tab  = (const float*)d_in[12];
    const float* fc1_w     = (const float*)d_in[13];
    const float* fc1_b     = (const float*)d_in[14];
    const float* fc2_w     = (const float*)d_in[15];
    const float* fc2_b     = (const float*)d_in[16];
    const int*   rel_index = (const int*)d_in[17];
    float* out = (float*)d_out;

    char* ws = (char*)d_ws;
    size_t off = 0;
    auto wsalloc = [&](size_t bytes) -> void* {
        void* p = ws + off;
        off += (bytes + 255) & ~(size_t)255;
        return p;
    };
    unsigned short* bqkv    = (unsigned short*)wsalloc(256 * 768  * 2);
    unsigned short* bproj   = (unsigned short*)wsalloc(256 * 256  * 2);
    unsigned short* b1w     = (unsigned short*)wsalloc(256 * 1024 * 2);
    unsigned short* b2w     = (unsigned short*)wsalloc(1024 * 256 * 2);
    float*          biasmat = (float*)wsalloc(HEADS * NTOK * NTOK * 4);
    float*          reg_upd = (float*)wsalloc((size_t)BATCH * NWIN * DIMC * 4);

    prep_kernel<<<256, 256, 0, stream>>>(qkv_w, proj_w, fc1_w, fc2_w, bias_tab,
                                         rel_index, bqkv, bproj, b1w, b2w, biasmat);
    rsa_kernel<<<BATCH, 256, 0, stream>>>(regional, n1g, n1b, bqkv, qkv_b,
                                          bproj, proj_b, reg_upd);
    lsa_kernel<<<NBLK, 256, 0, stream>>>(x, reg_upd, n2g, n2b, n3g, n3b,
                                         bqkv, qkv_b, bproj, proj_b,
                                         b1w, fc1_b, b2w, fc2_b, biasmat, out);
}
